// ConvolutionBase_in_out_6889127542990
// MI455X (gfx1250) — matrix operand order repaired and checked by reading, not compiled, not run on hardware
//
#include <hip/hip_runtime.h>
#include <hip/hip_bf16.h>
#include <math.h>

typedef __bf16 bf16_t;
typedef __attribute__((ext_vector_type(16))) __bf16 v16bf;
typedef __attribute__((ext_vector_type(8)))  __bf16 v8bf;
typedef __attribute__((ext_vector_type(8)))  float  v8f;

#define DFEAT 64
#define NLAB  8
#define INCH  256   // 4*DFEAT
#define OUTCH 64

// ---------------------------------------------------------------------------
// float atomic max via signed/unsigned integer ordering trick.
// Buffer must be initialized to -inf (0xFF800000).
// ---------------------------------------------------------------------------
__device__ __forceinline__ void atomicMaxF32(float* addr, float val) {
    if (val >= 0.0f) atomicMax((int*)addr,          __float_as_int(val));
    else             atomicMin((unsigned int*)addr, __float_as_uint(val));
}

// ---------------------------------------------------------------------------
// Workspace init: [ 4*ND sums = 0 | 4*ND maxes = -inf | 2*N counts = 0 ]
// ---------------------------------------------------------------------------
__global__ void init_ws_kernel(float* base, long nsum, long nmax, long ncnt) {
    long i      = (long)blockIdx.x * blockDim.x + threadIdx.x;
    long stride = (long)gridDim.x * blockDim.x;
    long total  = nsum + nmax + ncnt;
    const float ninf = __int_as_float(0xFF800000);
    for (; i < total; i += stride)
        base[i] = (i >= nsum && i < nsum + nmax) ? ninf : 0.0f;
}

// ---------------------------------------------------------------------------
// weight [256,64] f32 -> wT [64,256] bf16 (K-contiguous per output column,
// so a WMMA B-fragment is one contiguous 32B load per lane)
// ---------------------------------------------------------------------------
__global__ void prep_weight_kernel(const float* __restrict__ w, bf16_t* __restrict__ wT) {
    int i = blockIdx.x * blockDim.x + threadIdx.x;
    if (i >= INCH * OUTCH) return;
    int k = i / OUTCH, n = i % OUTCH;
    wT[n * INCH + k] = (bf16_t)w[i];
}

// ---------------------------------------------------------------------------
// Fused edge pass: one wave (32 lanes) per edge, 2 feature dims per lane.
//   ef = edge_label @ trans_weight   (16 FMAs/lane — cheaper than any WMMA
//                                     staging for a K=8 contraction)
//   att = sigmoid(lrelu((xr*xc)@aw) + lrelu(ef@eaw))  via shfl_xor reduction
//   scatter 4 sum streams + 4 max streams + degree counts with atomics.
// Memory bound: ~2.5KB traffic/edge dominates; VALU work is negligible.
// ---------------------------------------------------------------------------
__global__ void edge_kernel(const float* __restrict__ x,   const int* __restrict__ ei,
                            const float* __restrict__ elab, const float* __restrict__ tw,
                            const float* __restrict__ aw,   const float* __restrict__ eaw,
                            float* __restrict__ sums, float* __restrict__ maxs,
                            float* __restrict__ cntR, float* __restrict__ cntC,
                            int E, int Nn) {
    int wave = (int)(((long)blockIdx.x * blockDim.x + threadIdx.x) >> 5);
    if (wave >= E) return;
    int lane = threadIdx.x & 31;
    int e = wave;
    int row = ei[e];
    int col = ei[E + e];
    int d0 = lane * 2;

    const float2 xr = *(const float2*)(x + (long)row * DFEAT + d0);
    const float2 xc = *(const float2*)(x + (long)col * DFEAT + d0);

    float ef0 = 0.0f, ef1 = 0.0f;
#pragma unroll
    for (int l = 0; l < NLAB; ++l) {
        float lab = elab[(long)e * NLAB + l];          // wave-uniform, L2-broadcast
        ef0 = fmaf(lab, tw[l * DFEAT + d0],     ef0);
        ef1 = fmaf(lab, tw[l * DFEAT + d0 + 1], ef1);
    }

    float pA = xr.x * xc.x * aw[d0] + xr.y * xc.y * aw[d0 + 1];
    float pE = ef0 * eaw[d0] + ef1 * eaw[d0 + 1];
#pragma unroll
    for (int o = 16; o > 0; o >>= 1) {                 // wave32 butterfly reduce
        pA += __shfl_xor(pA, o, 32);
        pE += __shfl_xor(pE, o, 32);
    }
    float la = pA > 0.0f ? pA : 0.01f * pA;
    float le = pE > 0.0f ? pE : 0.01f * pE;
    float att = 1.0f / (1.0f + expf(-(la + le)));

    float wef0 = ef0  * att, wef1 = ef1  * att;        // weighted edge feats
    float wnf0 = xc.x * att, wnf1 = xc.y * att;        // weighted target
    float wsf0 = xr.x * att, wsf1 = xr.y * att;        // weighted source

    long ND = (long)Nn * DFEAT;
    long rb = (long)row * DFEAT + d0;
    long cb = (long)col * DFEAT + d0;

    // buffer 0: out         = agg(wnf, row)
    atomicAdd(sums + 0 * ND + rb,     wnf0);  atomicAdd(sums + 0 * ND + rb + 1, wnf1);
    atomicMaxF32(maxs + 0 * ND + rb,  wnf0);  atomicMaxF32(maxs + 0 * ND + rb + 1, wnf1);
    // buffer 1: opinion     = agg(wef, row)
    atomicAdd(sums + 1 * ND + rb,     wef0);  atomicAdd(sums + 1 * ND + rb + 1, wef1);
    atomicMaxF32(maxs + 1 * ND + rb,  wef0);  atomicMaxF32(maxs + 1 * ND + rb + 1, wef1);
    // buffer 2: inn         = agg(wsf, col)
    atomicAdd(sums + 2 * ND + cb,     wsf0);  atomicAdd(sums + 2 * ND + cb + 1, wsf1);
    atomicMaxF32(maxs + 2 * ND + cb,  wsf0);  atomicMaxF32(maxs + 2 * ND + cb + 1, wsf1);
    // buffer 3: inn_opinion = agg(wef, col)
    atomicAdd(sums + 3 * ND + cb,     wef0);  atomicAdd(sums + 3 * ND + cb + 1, wef1);
    atomicMaxF32(maxs + 3 * ND + cb,  wef0);  atomicMaxF32(maxs + 3 * ND + cb + 1, wef1);

    if (lane == 0) {
        atomicAdd(cntR + row, 1.0f);
        atomicAdd(cntC + col, 1.0f);
    }
}

// ---------------------------------------------------------------------------
// Finalize: mean*a0 + max*a1 + sum*a2 (softmax'd agg weights), empty-segment
// max -> 0, write cat[N,256] in bf16 for the WMMA GEMM.
// ---------------------------------------------------------------------------
__global__ void finalize_cat_kernel(const float* __restrict__ sums, const float* __restrict__ maxs,
                                    const float* __restrict__ cntR, const float* __restrict__ cntC,
                                    const float* __restrict__ aggw, bf16_t* __restrict__ cat,
                                    int Nn) {
    int idx = blockIdx.x * blockDim.x + threadIdx.x;
    int total = Nn * DFEAT;
    if (idx >= total) return;
    int n = idx / DFEAT, d = idx % DFEAT;

    float a0 = aggw[0], a1 = aggw[1], a2 = aggw[2];
    float mx3 = fmaxf(a0, fmaxf(a1, a2));
    float e0 = expf(a0 - mx3), e1 = expf(a1 - mx3), e2 = expf(a2 - mx3);
    float inv = 1.0f / (e0 + e1 + e2);
    a0 = e0 * inv; a1 = e1 * inv; a2 = e2 * inv;

    float cr = cntR[n], cc = cntC[n];
    long ND = (long)Nn * DFEAT;
#pragma unroll
    for (int b = 0; b < 4; ++b) {
        float cnt = (b < 2) ? cr : cc;
        float s = sums[(long)b * ND + idx];
        float m = maxs[(long)b * ND + idx];
        m = (cnt > 0.0f) ? m : 0.0f;
        float val = (s / fmaxf(cnt, 1.0f)) * a0 + m * a1 + s * a2;
        cat[(long)n * INCH + b * DFEAT + d] = (bf16_t)val;
    }
}

// ---------------------------------------------------------------------------
// Final GEMM: cat[N,256](bf16) @ wT'[256,64](bf16) + bias -> out[N,64](f32)
// One wave per 16-row tile; 4 accumulator tiles (full 64 cols) kept in VGPRs;
// K=256 -> 8 steps of v_wmma_f32_16x16x32_bf16 per tile = 32 WMMAs / wave.
// A-fragment layout (16-bit A 16x32): lane m=l%16, K = kc+16*hi .. +15
// (matches B) -> two contiguous 16B chunks per lane. B pre-transposed so each
// lane's fragment (N=l%16, K = kc+16*hi .. +15) is one contiguous 32B load.
// ---------------------------------------------------------------------------
__global__ void gemm_wmma_kernel(const bf16_t* __restrict__ cat, const bf16_t* __restrict__ wT,
                                 const float* __restrict__ bias, float* __restrict__ out,
                                 int Nn) {
    int wave = (int)(((long)blockIdx.x * blockDim.x + threadIdx.x) >> 5);
    int lane = threadIdx.x & 31;
    int m0 = wave * 16;
    if (m0 >= Nn) return;

    int mlow = lane & 15;
    int hi   = lane >> 4;
    const bf16_t* arow = cat + (long)(m0 + mlow) * INCH;

    v8f acc[4] = {};
#pragma unroll
    for (int ks = 0; ks < 8; ++ks) {
        int kc = ks * 32;
        v8bf aLo = *(const v8bf*)(arow + kc +      hi * 16);
        v8bf aHi = *(const v8bf*)(arow + kc + 8 + hi * 16);
        v16bf a;
#pragma unroll
        for (int i = 0; i < 8; ++i) { a[i] = aLo[i]; a[i + 8] = aHi[i]; }
#pragma unroll
        for (int t = 0; t < 4; ++t) {
            v16bf b = *(const v16bf*)(wT + (long)(t * 16 + mlow) * INCH + kc + hi * 16);
            acc[t] = __builtin_amdgcn_wmma_f32_16x16x32_bf16(
                false, a, false, b, (short)0, acc[t], false, false);
        }
    }

#pragma unroll
    for (int t = 0; t < 4; ++t) {
        int ncol = t * 16 + mlow;
        float bv = bias[ncol];
#pragma unroll
        for (int v = 0; v < 8; ++v) {
            int m = m0 + v + 8 * hi;               // C/D layout: M = vgpr + 8*(lane>=16)
            out[(long)m * OUTCH + ncol] = acc[t][v] + bv;
        }
    }
}

// ---------------------------------------------------------------------------
extern "C" void kernel_launch(void* const* d_in, const int* in_sizes, int n_in,
                              void* d_out, int out_size, void* d_ws, size_t ws_size,
                              hipStream_t stream) {
    const float* x    = (const float*)d_in[0];
    const int*   ei   = (const int*)  d_in[1];
    const float* elab = (const float*)d_in[2];
    const float* w    = (const float*)d_in[3];
    const float* bias = (const float*)d_in[4];
    const float* tw   = (const float*)d_in[5];
    const float* aw   = (const float*)d_in[6];
    const float* eaw  = (const float*)d_in[7];
    const float* aggw = (const float*)d_in[8];
    float* out = (float*)d_out;

    int Nn = in_sizes[0] / DFEAT;   // 100000
    int E  = in_sizes[1] / 2;       // 1000000
    long ND = (long)Nn * DFEAT;

    // workspace carve-up
    float*  sums = (float*)d_ws;            // 4*ND f32
    float*  maxs = sums + 4 * ND;           // 4*ND f32
    float*  cntR = maxs + 4 * ND;           // Nn f32
    float*  cntC = cntR + Nn;               // Nn f32
    bf16_t* wT   = (bf16_t*)(cntC + Nn);    // 256*64 bf16 (16B-aligned)
    bf16_t* cat  = wT + INCH * OUTCH;       // Nn*256 bf16 (16B-aligned)

    // 1. init sums/maxes/counts
    init_ws_kernel<<<2048, 256, 0, stream>>>(sums, 4 * ND, 4 * ND, 2L * Nn);

    // 2. weight -> bf16, transposed [64,256]
    prep_weight_kernel<<<(INCH * OUTCH + 255) / 256, 256, 0, stream>>>(w, wT);

    // 3. fused edge pass (one wave per edge)
    {
        long waves = E;
        long blocks = (waves * 32 + 255) / 256;
        edge_kernel<<<(int)blocks, 256, 0, stream>>>(x, ei, elab, tw, aw, eaw,
                                                     sums, maxs, cntR, cntC, E, Nn);
    }

    // 4. aggregate mix -> cat (bf16)
    finalize_cat_kernel<<<(int)((ND + 255) / 256), 256, 0, stream>>>(
        sums, maxs, cntR, cntC, aggw, cat, Nn);

    // 5. WMMA GEMM + bias -> out
    {
        int tiles  = (Nn + 15) / 16;                // waves
        int blocks = (tiles * 32 + 255) / 256;
        gemm_wmma_kernel<<<blocks, 256, 0, stream>>>(cat, wT, bias, out, Nn);
    }
    (void)n_in; (void)out_size; (void)ws_size;
}
